// MultiHeadSelfAttention_55697135895054
// MI455X (gfx1250) — matrix operand order repaired and checked by reading, not compiled, not run on hardware
//
#include <hip/hip_runtime.h>

#define TDIM 2048
#define DDIM 512
#define HNUM 8
#define BNUM 8

typedef __attribute__((ext_vector_type(16))) _Float16 v16h;
typedef __attribute__((ext_vector_type(8)))  _Float16 v8h;
typedef __attribute__((ext_vector_type(8)))  float    v8f;
typedef __attribute__((ext_vector_type(4)))  float    v4f;

__device__ __forceinline__ v8f wmma16(v16h a, v16h b, v8f c) {
  // D = A(16x32 f16) * B(32x16 f16) + C(16x16 f32)
  return __builtin_amdgcn_wmma_f32_16x16x32_f16(false, a, false, b, (short)0, c, false, false);
}

// A-operand as packed here (16x32, 16-bit, contiguous k, B uses the same): lanes 0-15 hold row=lane, K={0..15};
// lanes 16-31 hold row=lane-16, K={16..31}.
__device__ __forceinline__ v16h loadA_f16(const _Float16* base, int half) {
  v8h x0 = *(const v8h*)(base + half * 16);
  v8h x1 = *(const v8h*)(base + half * 16 + 8);
  v16h a;
#pragma unroll
  for (int i = 0; i < 8; ++i) { a[i] = x0[i]; a[8 + i] = x1[i]; }
  return a;
}

__device__ __forceinline__ v16h loadA_f32(const float* base, int half) {
  v4f x0 = ((const v4f*)(base + half * 16))[0];
  v4f x1 = ((const v4f*)(base + half * 16))[1];
  v4f y0 = ((const v4f*)(base + half * 16 + 8))[0];
  v4f y1 = ((const v4f*)(base + half * 16 + 8))[1];
  v16h a;
#pragma unroll
  for (int i = 0; i < 4; ++i) {
    a[i]      = (_Float16)x0[i];
    a[4 + i]  = (_Float16)x1[i];
    a[8 + i]  = (_Float16)y0[i];
    a[12 + i] = (_Float16)y1[i];
  }
  return a;
}

__global__ void cvt_f32_to_f16(const float* __restrict__ in, _Float16* __restrict__ out, int n) {
  int i = blockIdx.x * blockDim.x + threadIdx.x;
  if (i < n) out[i] = (_Float16)in[i];
}

// out[M=16384, N=512] = (A[M,512] @ W[N,512]^T + bias (+ A2 @ W2^T + bias2)) * oscale
// One wave computes a 16x64 tile. MODE: 0 = f16 row-major, 1 = f16 transposed
// per-head (Vt[((b*H+h)*64+d)*T + t]), 2 = f32 row-major.
template <bool A_F32, int MODE>
__global__ void __launch_bounds__(32, 1)
gemm512(const void* __restrict__ Av, const _Float16* __restrict__ W,
        const float* __restrict__ bias,
        const void* __restrict__ A2v, const _Float16* __restrict__ W2,
        const float* __restrict__ bias2,
        void* __restrict__ outv, float oscale) {
  const int lane = threadIdx.x;
  const int half = lane >> 4;
  const int lm   = lane & 15;
  const int m0 = blockIdx.x * 16;
  const int n0 = blockIdx.y * 64;

  v8f c[4] = {};
  const long arow = (long)(m0 + lm) * DDIM;

  for (int kk = 0; kk < DDIM; kk += 32) {
    v16h a;
    if (A_F32) a = loadA_f32((const float*)Av + arow + kk, half);
    else       a = loadA_f16((const _Float16*)Av + arow + kk, half);
#pragma unroll
    for (int t = 0; t < 4; ++t) {
      // B[k][n] = W[n][k]: lane holds column n = n0+t*16+lm, 16 contiguous k
      const _Float16* wp = W + (long)(n0 + t * 16 + lm) * DDIM + kk + half * 16;
      c[t] = wmma16(a, *(const v16h*)wp, c[t]);
    }
  }
  if (A2v != nullptr) {  // fused second GEMM (pos projection folded into KP)
    for (int kk = 0; kk < DDIM; kk += 32) {
      v16h a = loadA_f32((const float*)A2v + arow + kk, half);
#pragma unroll
      for (int t = 0; t < 4; ++t) {
        const _Float16* wp = W2 + (long)(n0 + t * 16 + lm) * DDIM + kk + half * 16;
        c[t] = wmma16(a, *(const v16h*)wp, c[t]);
      }
    }
  }

#pragma unroll
  for (int t = 0; t < 4; ++t) {
    const int n = n0 + t * 16 + lm;
    float bv = bias[n];
    if (bias2 != nullptr) bv += bias2[n];
#pragma unroll
    for (int r = 0; r < 8; ++r) {
      const int m = m0 + r + 8 * half;  // C layout: row r+8*(lane/16), col lane%16
      float val = (c[t][r] + bv) * oscale;
      if (MODE == 0) {
        ((_Float16*)outv)[(long)m * DDIM + n] = (_Float16)val;
      } else if (MODE == 1) {
        const int b_ = m / TDIM, tt = m % TDIM;
        const int h_ = n >> 6, d = n & 63;
        ((_Float16*)outv)[(long)((b_ * HNUM + h_) * 64 + d) * TDIM + tt] = (_Float16)val;
      } else {
        ((float*)outv)[(long)m * DDIM + n] = val;
      }
    }
  }
}

// Flash attention: one wave per (b, h, 32-query tile). Q is pre-scaled by
// 1/sqrt(64) in its projection, so scores = Qs @ KP^T directly. Online softmax
// with row-max via half-wave shuffles; row-sum via WMMA against an all-ones B.
__global__ void __launch_bounds__(32, 1)
flash_attn(const _Float16* __restrict__ Q, const _Float16* __restrict__ KP,
           const _Float16* __restrict__ Vt, _Float16* __restrict__ Ctx) {
  __shared__ _Float16 pbuf[2][16 * 32];
  const int lane = threadIdx.x;
  const int half = lane >> 4;
  const int lm   = lane & 15;
  const int q0 = blockIdx.x * 32;
  const int h  = blockIdx.y;
  const int b  = blockIdx.z;

  v16h qa[2][2];
#pragma unroll
  for (int qt = 0; qt < 2; ++qt) {
    const _Float16* qrow = Q + (long)(b * TDIM + q0 + qt * 16 + lm) * DDIM + h * 64;
    qa[qt][0] = loadA_f16(qrow, half);       // d = 0..31
    qa[qt][1] = loadA_f16(qrow + 32, half);  // d = 32..63
  }
  v16h ones;
#pragma unroll
  for (int i = 0; i < 16; ++i) ones[i] = (_Float16)1.0f;

  const _Float16* kpbase = KP + (long)b * TDIM * DDIM + h * 64;
  const _Float16* vtbase = Vt + (long)(b * HNUM + h) * 64 * TDIM;

  v8f acc[2][4] = {};
  float mrow[2][8], lrow[2][8];
#pragma unroll
  for (int qt = 0; qt < 2; ++qt)
#pragma unroll
    for (int r = 0; r < 8; ++r) { mrow[qt][r] = -1e30f; lrow[qt][r] = 0.f; }

  for (int j = 0; j < TDIM; j += 32) {
    // ---- scores: 32 queries x 32 keys, K-dim 64 ----
    v8f s[2][2] = {};
#pragma unroll
    for (int kt = 0; kt < 2; ++kt) {
      // one KP B-operand pair live at a time (16 VGPRs)
      const _Float16* kp = kpbase + (long)(j + kt * 16 + lm) * DDIM + half * 16;
      const v16h b0 = *(const v16h*)kp;
      const v16h b1 = *(const v16h*)(kp + 32);
#pragma unroll
      for (int qt = 0; qt < 2; ++qt) {
        s[qt][kt] = wmma16(qa[qt][0], b0, s[qt][kt]);
        s[qt][kt] = wmma16(qa[qt][1], b1, s[qt][kt]);
      }
    }

    // ---- online softmax per q-tile ----
#pragma unroll
    for (int qt = 0; qt < 2; ++qt) {
#pragma unroll
      for (int r = 0; r < 8; ++r) {
        float mx = fmaxf(s[qt][0][r], s[qt][1][r]);  // row reduce in 16-lane half
        mx = fmaxf(mx, __shfl_xor(mx, 1, 32));
        mx = fmaxf(mx, __shfl_xor(mx, 2, 32));
        mx = fmaxf(mx, __shfl_xor(mx, 4, 32));
        mx = fmaxf(mx, __shfl_xor(mx, 8, 32));
        const float mnew = fmaxf(mrow[qt][r], mx);
        const float sc = __expf(mrow[qt][r] - mnew);
        mrow[qt][r] = mnew;
        const float p0 = __expf(s[qt][0][r] - mnew);
        const float p1 = __expf(s[qt][1][r] - mnew);
        lrow[qt][r] *= sc;
        acc[qt][0][r] *= sc; acc[qt][1][r] *= sc;
        acc[qt][2][r] *= sc; acc[qt][3][r] *= sc;
        // C-layout -> A-layout bounce through LDS (f16)
        pbuf[qt][(r + 8 * half) * 32 + lm]      = (_Float16)p0;
        pbuf[qt][(r + 8 * half) * 32 + 16 + lm] = (_Float16)p1;
      }
    }
    __syncthreads();

    // ---- P as A-operands ----
    v16h pa[2];
#pragma unroll
    for (int qt = 0; qt < 2; ++qt) {
      v8h x0 = *(const v8h*)&pbuf[qt][lm * 32 + half * 16];
      v8h x1 = *(const v8h*)&pbuf[qt][lm * 32 + half * 16 + 8];
#pragma unroll
      for (int i = 0; i < 8; ++i) { pa[qt][i] = x0[i]; pa[qt][8 + i] = x1[i]; }
    }

    // row sums of P via WMMA with all-ones B (every column holds the sum)
#pragma unroll
    for (int qt = 0; qt < 2; ++qt) {
      v8f ssum = {};
      ssum = wmma16(pa[qt], ones, ssum);
#pragma unroll
      for (int r = 0; r < 8; ++r) lrow[qt][r] += ssum[r];
    }

    // ---- ctx += P @ V : load each Vt B-tile once, share across q-tiles ----
#pragma unroll
    for (int t = 0; t < 4; ++t) {
      const v16h bvt = *(const v16h*)(vtbase + (long)(t * 16 + lm) * TDIM + j + half * 16);
      acc[0][t] = wmma16(pa[0], bvt, acc[0][t]);
      acc[1][t] = wmma16(pa[1], bvt, acc[1][t]);
    }
    __syncthreads();
  }

#pragma unroll
  for (int qt = 0; qt < 2; ++qt)
#pragma unroll
    for (int r = 0; r < 8; ++r) {
      const float inv = 1.0f / lrow[qt][r];
      _Float16* crow = Ctx + (long)(b * TDIM + q0 + qt * 16 + r + 8 * half) * DDIM + h * 64;
#pragma unroll
      for (int t = 0; t < 4; ++t) crow[t * 16 + lm] = (_Float16)(acc[qt][t][r] * inv);
    }
}

extern "C" void kernel_launch(void* const* d_in, const int* in_sizes, int n_in,
                              void* d_out, int out_size, void* d_ws, size_t ws_size,
                              hipStream_t stream) {
  const float* x   = (const float*)d_in[0];
  const float* pos = (const float*)d_in[1];
  const float* Wq  = (const float*)d_in[2];
  const float* bq  = (const float*)d_in[3];
  const float* Wk  = (const float*)d_in[4];
  const float* bk  = (const float*)d_in[5];
  const float* Wv  = (const float*)d_in[6];
  const float* bv  = (const float*)d_in[7];
  const float* Wp  = (const float*)d_in[8];
  const float* bp  = (const float*)d_in[9];
  const float* Wo  = (const float*)d_in[10];
  const float* bo  = (const float*)d_in[11];

  const size_t WSZ  = (size_t)DDIM * DDIM;          // 262144 elements
  const size_t NBIG = (size_t)BNUM * TDIM * DDIM;   // 8388608 elements

  _Float16* p    = (_Float16*)d_ws;
  _Float16* Wq_h = p + 0 * WSZ;
  _Float16* Wk_h = p + 1 * WSZ;
  _Float16* Wv_h = p + 2 * WSZ;
  _Float16* Wp_h = p + 3 * WSZ;
  _Float16* Wo_h = p + 4 * WSZ;
  _Float16* Qh   = p + 5 * WSZ;
  _Float16* KPh  = Qh + 1 * NBIG;
  _Float16* Vth  = Qh + 2 * NBIG;
  _Float16* Ctx  = Qh + 3 * NBIG;

  // 1) weights f32 -> f16
  cvt_f32_to_f16<<<WSZ / 256, 256, 0, stream>>>(Wq, Wq_h, (int)WSZ);
  cvt_f32_to_f16<<<WSZ / 256, 256, 0, stream>>>(Wk, Wk_h, (int)WSZ);
  cvt_f32_to_f16<<<WSZ / 256, 256, 0, stream>>>(Wv, Wv_h, (int)WSZ);
  cvt_f32_to_f16<<<WSZ / 256, 256, 0, stream>>>(Wp, Wp_h, (int)WSZ);
  cvt_f32_to_f16<<<WSZ / 256, 256, 0, stream>>>(Wo, Wo_h, (int)WSZ);

  const dim3 gg((BNUM * TDIM) / 16, DDIM / 64);  // (1024, 8)

  // 2) projections (all WMMA). Q is pre-scaled by 1/sqrt(DH)=0.125 (exact pow2).
  //    KP = K + Pos fused as a dual-pass GEMM. V stored transposed per head.
  gemm512<true, 0><<<gg, 32, 0, stream>>>(x, Wq_h, bq, nullptr, nullptr, nullptr, Qh, 0.125f);
  gemm512<true, 0><<<gg, 32, 0, stream>>>(x, Wk_h, bk, pos, Wp_h, bp, KPh, 1.0f);
  gemm512<true, 1><<<gg, 32, 0, stream>>>(x, Wv_h, bv, nullptr, nullptr, nullptr, Vth, 1.0f);

  // 3) flash attention (32 queries per wave)
  flash_attn<<<dim3(TDIM / 32, HNUM, BNUM), 32, 0, stream>>>(Qh, KPh, Vth, Ctx);

  // 4) output projection -> f32 d_out
  gemm512<false, 2><<<gg, 32, 0, stream>>>(Ctx, Wo_h, bo, nullptr, nullptr, nullptr, d_out, 1.0f);
}
